// ConsciousAttention_73538430042407
// MI455X (gfx1250) — matrix operand order repaired and checked by reading, not compiled, not run on hardware
//
#include <hip/hip_runtime.h>
#include <hip/hip_bf16.h>

typedef __attribute__((ext_vector_type(16))) _Float16 v16h;
typedef __attribute__((ext_vector_type(8)))  float    v8f;

#define NROWS 500000
#define DDIM  256
#define HDIM  256
#define KTOP  64

__device__ __forceinline__ unsigned fenc(float f) {
  unsigned u = __float_as_uint(f);
  return (u & 0x80000000u) ? ~u : (u | 0x80000000u);
}
__device__ __forceinline__ float fdec(unsigned e) {
  unsigned u = (e & 0x80000000u) ? (e & 0x7fffffffu) : ~e;
  return __uint_as_float(u);
}

// ---------------------------------------------------------------------------
// K0: q = normalize(Wq@cb+bq)*2 ; wkq = Wk^T q /16 ; c0 = bk.q/16 ;
//     W1 -> f16 WMMA B-fragment layout ; zero accumulators (graph-replay safe)
// ---------------------------------------------------------------------------
__launch_bounds__(256)
__global__ void prep_kernel(const float* __restrict__ cb,
                            const float* __restrict__ Wq, const float* __restrict__ bq,
                            const float* __restrict__ Wk, const float* __restrict__ bk,
                            const float* __restrict__ W1,
                            float* __restrict__ wkq, float* __restrict__ c0s,
                            unsigned* __restrict__ maxenc, float* __restrict__ Ssum,
                            _Float16* __restrict__ w1f) {
  __shared__ float sq[256];
  __shared__ float red[256];
  int t = threadIdx.x;
  float acc = bq[t];
  for (int d = 0; d < DDIM; ++d) acc += Wq[t * DDIM + d] * cb[d];
  sq[t] = acc;
  red[t] = acc * acc;
  __syncthreads();
  for (int s = 128; s > 0; s >>= 1) { if (t < s) red[t] += red[t + s]; __syncthreads(); }
  float scl = 2.0f / fmaxf(sqrtf(red[0]), 1e-12f);
  __syncthreads();
  // wkq[d] = (sum_h Wk[h,d]*q[h]) * scl / sqrt(256)
  float w = 0.f;
  for (int h = 0; h < HDIM; ++h) w += Wk[h * DDIM + t] * sq[h];
  wkq[t] = w * scl * 0.0625f;
  // c0 = bk.q * scl / 16
  red[t] = bk[t] * sq[t];
  __syncthreads();
  for (int s = 128; s > 0; s >>= 1) { if (t < s) red[t] += red[t + s]; __syncthreads(); }
  if (t == 0) {
    *c0s = red[0] * scl * 0.0625f;
    *maxenc = 0u;   // < fenc of any finite float
    *Ssum = 0.f;
  }
  // W1[512,256] f32 -> f16 fragments: B(k,n)=W1[n,k]
  // frag layout: ((nt*8+ks)*32 + lane) * 16 halfs; lane 0-15: K=0-7,16-23; lane 16-31: K=8-15,24-31
  for (int p = t; p < 65536; p += 256) {
    int j    = p & 7;
    int lane = (p >> 3) & 31;
    int ks   = (p >> 8) & 7;
    int nt   = p >> 11;
    int n = nt * 16 + (lane & 15);
    int k = ks * 32 + (j < 4 ? 2 * j : 16 + 2 * (j - 4)) + ((lane < 16) ? 0 : 8);
    w1f[2 * p]     = (_Float16)W1[n * DDIM + k];
    w1f[2 * p + 1] = (_Float16)W1[n * DDIM + k + 1];
  }
}

// ---------------------------------------------------------------------------
// K1: fused — per wave: 16 rows. Load x once -> f16 A frags + f32 logit GEMV,
//     then 32 n-tiles x 8 WMMA(16x16x32 f16) for the MLP, relu*w2 accumulate.
// ---------------------------------------------------------------------------
__launch_bounds__(256)
__global__ void fused_mlp_kernel(const float* __restrict__ x,
                                 const float* __restrict__ b1,
                                 const float* __restrict__ W2,
                                 const float* __restrict__ b2,
                                 const float* __restrict__ wkq,
                                 const float* __restrict__ c0s,
                                 unsigned* __restrict__ maxenc,
                                 const v16h* __restrict__ w1f,
                                 float* __restrict__ logits,
                                 float* __restrict__ scores) {
  __shared__ float lwkq[256];
  __shared__ float smax[8];
  int tid = threadIdx.x;
  lwkq[tid] = wkq[tid];
  __syncthreads();

  int wave  = tid >> 5;
  int lane  = tid & 31;
  int rlane = lane & 15;
  bool hi   = lane >= 16;
  int rowbase = blockIdx.x * 128 + wave * 16;
  int row = rowbase + rlane; if (row > NROWS - 1) row = NROWS - 1;  // clamp; writes guarded
  const float* xrow = x + (size_t)row * DDIM;

  // A fragments (16x32 f16 per k-step, ISA layout) + fused logit partial
  v16h a[8];
  float logitp = 0.f;
#pragma unroll
  for (int ks = 0; ks < 8; ++ks) {
    v16h av;
#pragma unroll
    for (int j = 0; j < 8; ++j) {
      int k0 = ks * 32 + (j < 4 ? 2 * j : 16 + 2 * (j - 4)) + (hi ? 8 : 0);
      float2 xv = *(const float2*)(xrow + k0);
      float2 wv = *(const float2*)(&lwkq[k0]);
      logitp += xv.x * wv.x + xv.y * wv.y;
      av[2 * j]     = (_Float16)xv.x;
      av[2 * j + 1] = (_Float16)xv.y;
    }
    a[ks] = av;
  }

  // logit: lanes L and L+16 each hold half the dot product of row L
  float logit = logitp + __shfl_xor(logitp, 16, 32) + c0s[0];
  int orow = rowbase + rlane;
  if (!hi && orow < NROWS) logits[orow] = logit;
  float m = logit;
#pragma unroll
  for (int off = 1; off < 16; off <<= 1) m = fmaxf(m, __shfl_xor(m, off, 32));
  if (lane == 0) smax[wave] = m;

  // MLP: 32 hidden tiles of 16 cols; relu then weight by w2, per-lane accumulate
  float part[8] = {0.f, 0.f, 0.f, 0.f, 0.f, 0.f, 0.f, 0.f};
  for (int nt = 0; nt < 32; ++nt) {
    v8f c = {0.f, 0.f, 0.f, 0.f, 0.f, 0.f, 0.f, 0.f};
    const v16h* wfb = w1f + (size_t)(nt * 8) * 32 + lane;
#pragma unroll
    for (int ks = 0; ks < 8; ++ks) {
      v16h bv = wfb[ks * 32];
      c = __builtin_amdgcn_wmma_f32_16x16x32_f16(false, a[ks], false, bv,
                                                 (short)0, c, false, false);
    }
    int col = nt * 16 + rlane;
    float b1v = b1[col];
    float w2v = W2[col];
#pragma unroll
    for (int r = 0; r < 8; ++r) {
      float h = c[r] + b1v;
      h = h > 0.f ? h : 0.f;
      part[r] += h * w2v;
    }
  }
  // reduce across the 16 lanes of each half (cols), rows r / r+8
#pragma unroll
  for (int r = 0; r < 8; ++r) {
#pragma unroll
    for (int off = 1; off < 16; off <<= 1) part[r] += __shfl_xor(part[r], off, 32);
  }
  float b2v = b2[0];
  if (rlane == 0) {  // lanes 0 (rows 0-7) and 16 (rows 8-15)
    int rb = rowbase + (hi ? 8 : 0);
#pragma unroll
    for (int r = 0; r < 8; ++r) {
      int rr = rb + r;
      if (rr < NROWS) scores[rr] = 1.f / (1.f + __expf(-(part[r] + b2v)));
    }
  }
  __syncthreads();
  if (tid == 0) {
    float mm = smax[0];
    for (int i = 1; i < 8; ++i) mm = fmaxf(mm, smax[i]);
    atomicMax(maxenc, fenc(mm));
  }
}

// ---------------------------------------------------------------------------
// K2: t_i = exp(l_i - m) * v_i (softmax denom cancels under L1 norm); sum -> S
// ---------------------------------------------------------------------------
__launch_bounds__(256)
__global__ void expsum_kernel(float* __restrict__ larr,  // in: logits, out: t
                              const float* __restrict__ scores,
                              const unsigned* __restrict__ maxenc,
                              float* __restrict__ Ssum) {
  __shared__ float red[256];
  int i = blockIdx.x * 256 + threadIdx.x;
  float m = fdec(*maxenc);
  float t = 0.f;
  if (i < NROWS) {
    t = __expf(larr[i] - m) * scores[i];
    larr[i] = t;
  }
  red[threadIdx.x] = t;
  __syncthreads();
  for (int s = 128; s > 0; s >>= 1) { if (threadIdx.x < s) red[threadIdx.x] += red[threadIdx.x + s]; __syncthreads(); }
  if (threadIdx.x == 0) atomicAdd(Ssum, red[0]);
}

// ---------------------------------------------------------------------------
// K3: allocation = t/S ; build sort keys (enc(t)<<32 | ~i)  [tie: lowest idx]
// ---------------------------------------------------------------------------
__launch_bounds__(256)
__global__ void alloc_keys_kernel(const float* __restrict__ tarr,
                                  const float* __restrict__ Ssum,
                                  float* __restrict__ alloc,
                                  unsigned long long* __restrict__ keys) {
  int i = blockIdx.x * 256 + threadIdx.x;
  if (i < NROWS) {
    float t = tarr[i];
    float S = fmaxf(*Ssum, 1e-12f);
    alloc[i] = t / S;
    keys[i] = ((unsigned long long)fenc(t) << 32) | (unsigned)(~(unsigned)i);
  }
}

// ---------------------------------------------------------------------------
// bitonic sort (descending) in LDS, NSORT power of two
// ---------------------------------------------------------------------------
template <int NSORT>
__device__ void bitonic_desc(unsigned long long* s) {
  for (int size = 2; size <= NSORT; size <<= 1) {
    for (int stride = size >> 1; stride > 0; stride >>= 1) {
      __syncthreads();
      for (int i = threadIdx.x; i < NSORT; i += blockDim.x) {
        int j = i ^ stride;
        if (j > i) {
          bool desc = ((i & size) == 0);
          unsigned long long A = s[i], B = s[j];
          if ((A < B) == desc) { s[i] = B; s[j] = A; }
        }
      }
    }
  }
  __syncthreads();
}

// K4/K5: keep local top-64 per 1024-element chunk (exact tournament)
__launch_bounds__(256)
__global__ void topk_reduce(const unsigned long long* __restrict__ in, int count,
                            unsigned long long* __restrict__ out) {
  __shared__ unsigned long long s[1024];
  int base = blockIdx.x * 1024;
  for (int i = threadIdx.x; i < 1024; i += 256) {
    int g = base + i;
    s[i] = (g < count) ? in[g] : 0ull;
  }
  bitonic_desc<1024>(s);
  if (threadIdx.x < 64) out[blockIdx.x * 64 + threadIdx.x] = s[threadIdx.x];
}

// K6: final sort of 2048 candidates; top-64 descending; fused row gather
__launch_bounds__(256)
__global__ void topk_final_gather(const unsigned long long* __restrict__ cand,
                                  const float* __restrict__ x,
                                  float* __restrict__ attended) {
  __shared__ unsigned long long s[2048];
  for (int i = threadIdx.x; i < 2048; i += 256) s[i] = cand[i];
  bitonic_desc<2048>(s);
  for (int e = threadIdx.x; e < KTOP * DDIM; e += 256) {
    int r = e >> 8;
    unsigned rowid = ~(unsigned)(s[r] & 0xffffffffull);
    attended[e] = x[(size_t)rowid * DDIM + (e & 255)];
  }
}

// ---------------------------------------------------------------------------
extern "C" void kernel_launch(void* const* d_in, const int* in_sizes, int n_in,
                              void* d_out, int out_size, void* d_ws, size_t ws_size,
                              hipStream_t stream) {
  const float* x  = (const float*)d_in[0];
  const float* cb = (const float*)d_in[1];
  const float* Wq = (const float*)d_in[2];
  const float* bq = (const float*)d_in[3];
  const float* Wk = (const float*)d_in[4];
  const float* bk = (const float*)d_in[5];
  const float* W1 = (const float*)d_in[6];
  const float* b1 = (const float*)d_in[7];
  const float* W2 = (const float*)d_in[8];
  const float* b2 = (const float*)d_in[9];

  float* out_att   = (float*)d_out;            // [64,256]
  float* out_alloc = out_att + KTOP * DDIM;    // [N]
  float* out_score = out_alloc + NROWS;        // [N]

  char* ws = (char*)d_ws;
  float*    wkq    = (float*)ws;               // 256 f32
  float*    c0s    = wkq + 256;
  unsigned* maxenc = (unsigned*)(wkq + 257);
  float*    Ssum   = wkq + 258;
  float*    logits = (float*)(ws + 2048);                                // N f32 (reused as t)
  unsigned long long* keys  = (unsigned long long*)(ws + 2048 + (size_t)NROWS * 4);
  unsigned long long* cand1 = keys + NROWS;    // 512*64 = 32768
  unsigned long long* cand2 = cand1 + 32768;   // 32*64  = 2048
  v16h* w1f = (v16h*)(cand2 + 2048);           // 512*256 f16 fragments (256 KB)

  prep_kernel<<<1, 256, 0, stream>>>(cb, Wq, bq, Wk, bk, W1,
                                     wkq, c0s, maxenc, Ssum, (_Float16*)w1f);
  fused_mlp_kernel<<<(NROWS + 127) / 128, 256, 0, stream>>>(
      x, b1, W2, b2, wkq, c0s, maxenc, w1f, logits, out_score);
  expsum_kernel<<<(NROWS + 255) / 256, 256, 0, stream>>>(logits, out_score, maxenc, Ssum);
  alloc_keys_kernel<<<(NROWS + 255) / 256, 256, 0, stream>>>(logits, Ssum, out_alloc, keys);
  topk_reduce<<<512, 256, 0, stream>>>(keys, NROWS, cand1);
  topk_reduce<<<32, 256, 0, stream>>>(cand1, 32768, cand2);
  topk_final_gather<<<1, 256, 0, stream>>>(cand2, x, out_att);
}
